// MoChA_9655086482078
// MI455X (gfx1250) — compile-verified, matrix operand order checked by reading, not run on hardware
//
#include <hip/hip_runtime.h>
#include <hip/hip_bf16.h>
#include <math.h>

// ---------- CDNA5 WMMA types ----------
typedef __attribute__((ext_vector_type(16))) __bf16 v16bf;
typedef __attribute__((ext_vector_type(8)))  __bf16 v8bf;
typedef __attribute__((ext_vector_type(8)))  float  v8f;

#define NEG_INF (-3.4028234663852886e38f)
#define MEPS 1e-6f

// Problem constants
constexpr int CB  = 16;     // batch
constexpr int CQ  = 128;    // queries
constexpr int CK  = 1500;   // keys
constexpr int CKP = 1504;   // padded keys (mult of 32)
constexpr int CD  = 512;    // model / attention dim
constexpr int CHM = 4;      // monotonic heads

// =====================================================================
// f32 -> bf16 with row padding (works for pad==0 too)
// =====================================================================
__global__ void k_pad_convert(const float* __restrict__ in, __bf16* __restrict__ out,
                              long rowsIn, long rowsPad, long cols, long batches) {
    long idx = (long)blockIdx.x * blockDim.x + threadIdx.x;
    long tot = batches * rowsPad * cols;
    if (idx >= tot) return;
    long d = idx % cols;
    long r = idx / cols;
    long k = r % rowsPad;
    long b = r / rowsPad;
    float v = (k < rowsIn) ? in[(b * rowsIn + k) * cols + d] : 0.0f;
    out[idx] = (__bf16)v;
}

// W[512,512] row-major -> Wt[n][k] bf16 (so per-output-column k-runs are contiguous)
__global__ void k_transpose_w(const float* __restrict__ W, __bf16* __restrict__ Wt) {
    int idx = blockIdx.x * blockDim.x + threadIdx.x; // 512*512
    int k = idx & 511;
    int n = idx >> 9;
    Wt[idx] = (__bf16)W[k * 512 + n];
}

// v_b[b, k, h*128+d] -> vt[b, h, d, k]  (so cv-GEMM B operand has k contiguous)
__global__ void k_transpose_v(const __bf16* __restrict__ vb, __bf16* __restrict__ vt) {
    long idx = (long)blockIdx.x * blockDim.x + threadIdx.x;
    long tot = (long)CB * CD * CKP;
    if (idx >= tot) return;
    long k  = idx % CKP;
    long r  = idx / CKP;
    long dh = r % CD;     // = h*128 + d (contiguous enumeration matches [h][d])
    long b  = r / CD;
    vt[idx] = vb[(b * CKP + k) * CD + dh];
}

// =====================================================================
// Generic batched WMMA GEMM:  C[M,N] = A[M,Kd] * Bt[N,Kd]^T  (+ epilogue)
//   A row-major with reduction dim contiguous; Bt row n holds column n of B.
//   128 threads = 4 waves, one 16x16 tile per wave.
// epilogue modes:
//   0: bf16 store, += bias[n]
//   1: f32  store, += bias[n]
//   2: f32  store of sigmoid(c*scale + r[0]) with mask, 0 beyond validN  (p_choose)
//   3: f32  store of c*scale with mask -> NEG_INF, NEG_INF beyond validN (e_chunk)
// =====================================================================
__global__ void k_wmma_gemm(
    const __bf16* __restrict__ A,  long aBatch, long aHead, int lda,
    const __bf16* __restrict__ Bt, long bBatch, long bHead, int ldb,
    void* __restrict__ Out,        long oBatch, long oHead, int ldo,
    const float* __restrict__ bias, const float* __restrict__ rptr,
    const int* __restrict__ maskp,
    int heads, int M, int N, int Kd, int mode, float scale, int validN) {

    const int lane = threadIdx.x & 31;
    const int wave = threadIdx.x >> 5;
    const int nt = N >> 4, mt = M >> 4;
    const long tile = (long)blockIdx.x * 4 + wave;
    if (tile >= (long)mt * nt) return;
    const int mtile = (int)(tile / nt);
    const int ntile = (int)(tile % nt);
    const int batch = blockIdx.y;
    const int b = batch / heads;
    const int h = batch % heads;

    const int hs = lane >> 4;      // half-select (K split per ISA layout)
    const int l15 = lane & 15;     // row (A) / col (B)

    const __bf16* arow = A  + (long)b * aBatch + (long)h * aHead + (long)(mtile * 16 + l15) * lda;
    const __bf16* brow = Bt + (long)b * bBatch + (long)h * bHead + (long)(ntile * 16 + l15) * ldb;

    v8f acc = {};
    for (int kk = 0; kk < Kd; kk += 32) {
        __builtin_prefetch(arow + kk + 256, 0, 1);
        const v8bf a0 = *(const v8bf*)(arow + kk + 8 * hs);
        const v8bf a1 = *(const v8bf*)(arow + kk + 16 + 8 * hs);
        const v8bf b0 = *(const v8bf*)(brow + kk + 8 * hs);
        const v8bf b1 = *(const v8bf*)(brow + kk + 16 + 8 * hs);
        v16bf af, bf_;
#pragma unroll
        for (int i = 0; i < 8; ++i) {
            af[i] = a0[i]; af[8 + i] = a1[i];
            bf_[i] = b0[i]; bf_[8 + i] = b1[i];
        }
        acc = __builtin_amdgcn_wmma_f32_16x16x32_bf16(
            false, af, false, bf_, (short)0, acc, false, false);
    }

    const long obase = (long)b * oBatch + (long)h * oHead;
    const int ncol = ntile * 16 + l15;
#pragma unroll
    for (int r2 = 0; r2 < 8; ++r2) {
        const int m = mtile * 16 + r2 + 8 * hs;       // C/D layout: vgpr r -> M=r / r+8
        const float cval = acc[r2];
        const long oidx = obase + (long)m * ldo + ncol;
        if (mode == 0) {
            float v = cval + (bias ? bias[ncol] : 0.0f);
            ((__bf16*)Out)[oidx] = (__bf16)v;
        } else if (mode == 1) {
            ((float*)Out)[oidx] = cval + (bias ? bias[ncol] : 0.0f);
        } else if (mode == 2) {
            bool ok = (ncol < validN);
            if (ok && maskp) ok = maskp[((long)b * M + m) * validN + ncol] != 0;
            float e = cval * scale + rptr[0];
            ((float*)Out)[oidx] = ok ? 1.0f / (1.0f + __expf(-e)) : 0.0f;
        } else {
            bool ok = (ncol < validN);
            if (ok && maskp) ok = maskp[((long)b * M + m) * validN + ncol] != 0;
            ((float*)Out)[oidx] = ok ? cval * scale : NEG_INF;
        }
    }
}

// =====================================================================
// Monotonic recurrence: one workgroup per (b,h), 512 threads x 3 items.
// Per q: safe_cumprod (exclusive scan of log(1-p)) then
//        a = p*cp*cumsum(a_prev / clip(cp,eps,1)).
// alpha may alias p (row q read fully before row q written).
// =====================================================================
__device__ __forceinline__ float block_excl_scan(float s, float* lds, int t) {
    lds[t] = s;
    __syncthreads();
    for (int off = 1; off < 512; off <<= 1) {
        float v = (t >= off) ? lds[t - off] : 0.0f;
        __syncthreads();
        lds[t] += v;
        __syncthreads();
    }
    float incl = lds[t];
    __syncthreads();
    return incl - s;
}

__global__ void k_recurrence(const float* __restrict__ p_all, float* __restrict__ alpha) {
    const int bh = blockIdx.x;           // 0..63  (b*4+h)
    const int t  = threadIdx.x;          // 0..511
    constexpr int IT = 3;                // 512*3 = 1536 >= 1504
    __shared__ float s_scan[512];

    float aprev[IT];
#pragma unroll
    for (int i = 0; i < IT; ++i) {
        int k = t * IT + i;
        aprev[i] = (k == 0) ? 1.0f : 0.0f;
    }
    const float* pbase = p_all + (long)bh * CQ * CKP;
    float*       abase = alpha + (long)bh * CQ * CKP;

    for (int q = 0; q < CQ; ++q) {
        float p[IT], lg[IT];
#pragma unroll
        for (int i = 0; i < IT; ++i) {
            int k = t * IT + i;
            p[i]  = (k < CKP) ? pbase[(long)q * CKP + k] : 0.0f;
            lg[i] = (k < CK) ? __logf(fminf(fmaxf(1.0f - p[i], MEPS), 1.0f)) : 0.0f;
        }
        // exclusive cumsum of log(1-p) -> cumprod
        float ex = block_excl_scan(lg[0] + lg[1] + lg[2], s_scan, t);
        float cp[IT], run = ex;
#pragma unroll
        for (int i = 0; i < IT; ++i) { cp[i] = __expf(run); run += lg[i]; }
        // inclusive cumsum of a_prev / clip(cp)
        float y[IT];
#pragma unroll
        for (int i = 0; i < IT; ++i) y[i] = aprev[i] / fminf(fmaxf(cp[i], MEPS), 1.0f);
        float ex2 = block_excl_scan(y[0] + y[1] + y[2], s_scan, t);
        run = ex2;
#pragma unroll
        for (int i = 0; i < IT; ++i) {
            run += y[i];
            int k = t * IT + i;
            float a = p[i] * cp[i] * run;
            if (k >= CK) a = 0.0f;
            if (k < CKP) abase[(long)q * CKP + k] = a;
            aprev[i] = a;
        }
        __syncthreads();
    }
}

// =====================================================================
// Chunkwise attention: per (b,q) row -> beta (bf16, padded with zeros).
//   sexp = clip(exp(e - max), 1e-5); denom = movsum(sexp, back=7)
//   beta_h = sexp * movsum(alpha_h/denom, fwd=7)
// =====================================================================
__global__ void k_beta(const float* __restrict__ e_chunk, const float* __restrict__ alpha,
                       __bf16* __restrict__ beta) {
    const int q = blockIdx.x;   // 0..127
    const int b = blockIdx.y;   // 0..15
    const int t = threadIdx.x;  // 0..255
    __shared__ float s_exp[CKP];
    __shared__ float s_den[CKP];
    __shared__ float s_t[CKP];
    __shared__ float s_red[256];

    const float* e = e_chunk + ((long)b * CQ + q) * CKP;

    float mx = NEG_INF;
    for (int k = t; k < CK; k += 256) mx = fmaxf(mx, e[k]);
    s_red[t] = mx;
    __syncthreads();
    for (int off = 128; off > 0; off >>= 1) {
        if (t < off) s_red[t] = fmaxf(s_red[t], s_red[t + off]);
        __syncthreads();
    }
    mx = s_red[0];
    __syncthreads();

    for (int k = t; k < CKP; k += 256)
        s_exp[k] = (k < CK) ? fmaxf(__expf(e[k] - mx), 1e-5f) : 0.0f;
    __syncthreads();

    for (int k = t; k < CKP; k += 256) {
        float d = 0.0f;
#pragma unroll
        for (int j = 0; j < 8; ++j) { int kk = k - j; if (kk >= 0) d += s_exp[kk]; }
        s_den[k] = d;
    }
    __syncthreads();

    for (int h = 0; h < CHM; ++h) {
        const float* al = alpha + (((long)b * CHM + h) * CQ + q) * CKP;
        __bf16* bo = beta + (((long)b * CHM + h) * CQ + q) * CKP;
        for (int k = t; k < CKP; k += 256)
            s_t[k] = (k < CK) ? al[k] / s_den[k] : 0.0f;   // SHARP == 1.0
        __syncthreads();
        for (int k = t; k < CKP; k += 256) {
            float m2 = 0.0f;
#pragma unroll
            for (int j = 0; j < 8; ++j) { int kk = k + j; if (kk < CK) m2 += s_t[kk]; }
            float bv = (k < CK) ? s_exp[k] * m2 : 0.0f;
            bo[k] = (__bf16)bv;
        }
        __syncthreads();
    }
}

// =====================================================================
// Host side
// =====================================================================
static inline unsigned cdiv(long a, long b) { return (unsigned)((a + b - 1) / b); }

extern "C" void kernel_launch(void* const* d_in, const int* in_sizes, int n_in,
                              void* d_out, int out_size, void* d_ws, size_t ws_size,
                              hipStream_t stream) {
    (void)in_sizes; (void)n_in; (void)out_size; (void)ws_size;

    const float* key   = (const float*)d_in[0];
    const float* query = (const float*)d_in[1];
    const float* value = (const float*)d_in[2];
    const int*   mask  = (const int*)  d_in[3];
    const float* Wk_m  = (const float*)d_in[4];
    const float* bk_m  = (const float*)d_in[5];
    const float* Wq_m  = (const float*)d_in[6];
    const float* bq_m  = (const float*)d_in[7];
    const float* rptr  = (const float*)d_in[8];
    const float* Wk_c  = (const float*)d_in[9];
    const float* bk_c  = (const float*)d_in[10];
    const float* Wq_c  = (const float*)d_in[11];
    const float* bq_c  = (const float*)d_in[12];
    const float* Wv    = (const float*)d_in[13];
    const float* bv    = (const float*)d_in[14];
    const float* Wout  = (const float*)d_in[15];
    const float* bout  = (const float*)d_in[16];

    // ---- workspace carve (256B aligned) ----
    size_t off = 0;
    auto carve = [&](size_t bytes) -> char* {
        char* p = (char*)d_ws + off;
        off += (bytes + 255) / 256 * 256;
        return p;
    };
    const size_t szKeyB = (size_t)CB * CKP * CD * 2;   // bf16 [B,KP,512]
    const size_t szQryB = (size_t)CB * CQ  * CD * 2;   // bf16 [B,128,512]
    const size_t szW    = (size_t)CD * CD * 2;

    __bf16* keyb = (__bf16*)carve(szKeyB);
    __bf16* valb = (__bf16*)carve(szKeyB);
    __bf16* qryb = (__bf16*)carve(szQryB);
    __bf16* wt_km = (__bf16*)carve(szW);
    __bf16* wt_qm = (__bf16*)carve(szW);
    __bf16* wt_kc = (__bf16*)carve(szW);
    __bf16* wt_qc = (__bf16*)carve(szW);
    __bf16* wt_v  = (__bf16*)carve(szW);
    __bf16* wt_o  = (__bf16*)carve(szW);
    __bf16* km_b = (__bf16*)carve(szKeyB);
    __bf16* kc_b = (__bf16*)carve(szKeyB);
    __bf16* v_b  = (__bf16*)carve(szKeyB);
    __bf16* qm_b = (__bf16*)carve(szQryB);
    __bf16* qc_b = (__bf16*)carve(szQryB);
    float*  palpha = (float*)carve((size_t)CB * CHM * CQ * CKP * 4); // p_choose, then alpha (aliased)
    float*  echk   = (float*)carve((size_t)CB * CQ * CKP * 4);
    __bf16* cvb    = (__bf16*)carve(szQryB);
    __bf16* vt   = keyb;   // reuse (keyb dead after kc GEMM); same byte size
    __bf16* beta = valb;   // reuse (valb dead after v GEMM); same byte size

    // ---- stage 0: convert / pad / transpose weights ----
    {
        long tot = (long)CB * CKP * CD;
        k_pad_convert<<<cdiv(tot, 256), 256, 0, stream>>>(key,   keyb, CK, CKP, CD, CB);
        k_pad_convert<<<cdiv(tot, 256), 256, 0, stream>>>(value, valb, CK, CKP, CD, CB);
        long totq = (long)CB * CQ * CD;
        k_pad_convert<<<cdiv(totq, 256), 256, 0, stream>>>(query, qryb, CQ, CQ, CD, CB);
        k_transpose_w<<<1024, 256, 0, stream>>>(Wk_m, wt_km);
        k_transpose_w<<<1024, 256, 0, stream>>>(Wq_m, wt_qm);
        k_transpose_w<<<1024, 256, 0, stream>>>(Wk_c, wt_kc);
        k_transpose_w<<<1024, 256, 0, stream>>>(Wq_c, wt_qc);
        k_transpose_w<<<1024, 256, 0, stream>>>(Wv,   wt_v);
        k_transpose_w<<<1024, 256, 0, stream>>>(Wout, wt_o);
    }

    auto gemm = [&](const __bf16* A, long aB, long aH, int lda,
                    const __bf16* Bt, long bB, long bH, int ldb,
                    void* Out, long oB, long oH, int ldo,
                    const float* bias, const float* rp, const int* mp,
                    int heads, int batches, int M, int N, int Kd,
                    int mode, float scale, int validN) {
        long tiles = (long)(M / 16) * (N / 16);
        dim3 g(cdiv(tiles, 4), (unsigned)batches);
        k_wmma_gemm<<<g, 128, 0, stream>>>(A, aB, aH, lda, Bt, bB, bH, ldb,
                                           Out, oB, oH, ldo, bias, rp, mp,
                                           heads, M, N, Kd, mode, scale, validN);
    };

    const long MKEY = (long)CB * CKP;   // 24064
    const long MQ   = (long)CB * CQ;    // 2048
    const float iscale = 0.044194173824159216f; // 1/sqrt(512)

    // ---- stage 1: projections (bf16 out, +bias) ----
    gemm(keyb, 0,0,CD, wt_km,0,0,CD, km_b, 0,0,CD, bk_m, nullptr,nullptr, 1,1, (int)MKEY, CD, CD, 0, 1.f, CD);
    gemm(keyb, 0,0,CD, wt_kc,0,0,CD, kc_b, 0,0,CD, bk_c, nullptr,nullptr, 1,1, (int)MKEY, CD, CD, 0, 1.f, CD);
    gemm(valb, 0,0,CD, wt_v, 0,0,CD, v_b,  0,0,CD, bv,   nullptr,nullptr, 1,1, (int)MKEY, CD, CD, 0, 1.f, CD);
    gemm(qryb, 0,0,CD, wt_qm,0,0,CD, qm_b, 0,0,CD, bq_m, nullptr,nullptr, 1,1, (int)MQ,   CD, CD, 0, 1.f, CD);
    gemm(qryb, 0,0,CD, wt_qc,0,0,CD, qc_b, 0,0,CD, bq_c, nullptr,nullptr, 1,1, (int)MQ,   CD, CD, 0, 1.f, CD);

    // ---- v transpose for cv GEMM (overwrites keyb region) ----
    {
        long tot = (long)CB * CD * CKP;
        k_transpose_v<<<cdiv(tot, 256), 256, 0, stream>>>(v_b, vt);
    }

    // ---- stage 2: scores ----
    // p_choose = sigmoid(qm.km^T/scale + r), masked   [B,4,Q,KP] f32
    gemm(qm_b, (long)CQ*CD, 128, CD,
         km_b, (long)CKP*CD, 128, CD,
         palpha, (long)CHM*CQ*CKP, (long)CQ*CKP, CKP,
         nullptr, rptr, mask, CHM, CB*CHM, CQ, CKP, 128, 2, iscale, CK);
    // e_chunk = qc.kc^T/scale, masked->NEG_INF        [B,Q,KP] f32
    gemm(qc_b, (long)CQ*CD, 0, CD,
         kc_b, (long)CKP*CD, 0, CD,
         echk, (long)CQ*CKP, 0, CKP,
         nullptr, nullptr, mask, 1, CB, CQ, CKP, CD, 3, iscale, CK);

    // ---- stage 3: monotonic recurrence (alpha overwrites p_choose in place) ----
    k_recurrence<<<CB * CHM, 512, 0, stream>>>(palpha, palpha);

    // ---- stage 4: chunkwise beta (bf16, overwrites valb region) ----
    {
        dim3 g(CQ, CB);
        k_beta<<<g, 256, 0, stream>>>(echk, palpha, beta);
    }

    // ---- stage 5: cv = beta @ v  per (b,h) -> cvb bf16 [B,Q,512] ----
    gemm(beta, (long)CHM*CQ*CKP, (long)CQ*CKP, CKP,
         vt,   (long)CHM*128*CKP, (long)128*CKP, CKP,
         cvb,  (long)CQ*CD, 128, CD,
         nullptr, nullptr, nullptr, CHM, CB*CHM, CQ, 128, CKP, 0, 1.f, 128);

    // ---- stage 6: out = cv @ Wout + bout  (f32 -> d_out) ----
    gemm(cvb, 0,0,CD, wt_o,0,0,CD, (float*)d_out, 0,0,CD, bout, nullptr,nullptr,
         1,1, (int)MQ, CD, CD, 1, 1.f, CD);
}
